// NSM_6227702579422
// MI455X (gfx1250) — compile-verified, matrix operand order checked by reading, not run on hardware
//
#include <hip/hip_runtime.h>
#include <hip/hip_bf16.h>
#include <stdint.h>

typedef __attribute__((ext_vector_type(16))) _Float16 v16h;
typedef __attribute__((ext_vector_type(8)))  float    v8f;
typedef __attribute__((ext_vector_type(4)))  int      v4i;

#define WMMA(a,b,c) __builtin_amdgcn_wmma_f32_16x16x32_f16(false,(a),false,(b),(short)0,(c),false,false)

constexpr int B_=256, L_=32, D_=128, V_=2000, P_=8, NPG_=128, EPN_=8;
constexpr int N_=B_*NPG_, E_=N_*EPN_, STEPS_=8, OUT_=1024;
#define NEGV (-1.0e30f)

// ---- fragment helpers (wave32, 16x16x32 f16 WMMA) -------------------------
// A/B fragment from f16 (LDS or global), row-major [fragrow][k]
__device__ __forceinline__ v16h fragA_s(const _Float16* p0, int ld, int lane){
  int r=lane&15, sel=(lane>>4)&1;
  const _Float16* p = p0 + r*ld + sel*8;
  v16h f;
  #pragma unroll
  for (int j=0;j<8;++j){ f[j]=p[j]; f[j+8]=p[j+16]; }
  return f;
}
// B fragment from f16 stored [k][fragrow] (transposed gather)
__device__ __forceinline__ v16h fragB_th(const _Float16* p0, int ld, int lane){
  int r=lane&15, sel=(lane>>4)&1;
  v16h f;
  #pragma unroll
  for (int j=0;j<8;++j){
    f[j]  =p0[(sel*8+j)*ld + r];
    f[j+8]=p0[(16+sel*8+j)*ld + r];
  }
  return f;
}
// ext rows (vocab_h has 2001 valid rows; zero-pad beyond)
__device__ __forceinline__ v16h frag_ext(const _Float16* vocab_h, int v0, int k0, int lane){
  int r=lane&15, sel=(lane>>4)&1;
  int v = v0 + r;
  const _Float16* p = (v <= V_) ? (vocab_h + v*D_ + k0 + sel*8) : nullptr;
  v16h f;
  #pragma unroll
  for (int j=0;j<8;++j){
    f[j]   = p ? p[j]    : (_Float16)0.f;
    f[j+8] = p ? p[j+16] : (_Float16)0.f;
  }
  return f;
}
// B fragment: rows = out dim d (fixed per lane), k = vocab index (guarded < V)
__device__ __forceinline__ v16h frag_vocT(const _Float16* vocab_h, int vbase, int dcol, int lane){
  int sel=(lane>>4)&1;
  v16h f;
  #pragma unroll
  for (int j=0;j<8;++j){
    int v1 = vbase + sel*8 + j;
    int v2 = v1 + 16;
    f[j]   = (v1 < V_) ? vocab_h[v1*D_ + dcol] : (_Float16)0.f;
    f[j+8] = (v2 < V_) ? vocab_h[v2*D_ + dcol] : (_Float16)0.f;
  }
  return f;
}

__device__ __forceinline__ float redmax16(float v){
  #pragma unroll
  for (int m=1;m<16;m<<=1) v = fmaxf(v, __shfl_xor(v,m,32));
  return v;
}
__device__ __forceinline__ float redsum16(float v){
  #pragma unroll
  for (int m=1;m<16;m<<=1) v += __shfl_xor(v,m,32);
  return v;
}
__device__ __forceinline__ float sigf(float x){ return 1.f/(1.f+__expf(-x)); }
__device__ __forceinline__ float eluf(float x){ return x>0.f ? x : expm1f(x); }

// ---- gfx1250 async global->LDS copy (16B per lane), guarded ---------------
__device__ __forceinline__ void lds_copy16_async(_Float16* dst, const _Float16* src){
#if defined(__gfx1250__) && __has_builtin(__builtin_amdgcn_global_load_async_to_lds_b128)
  // param0: global v4i*, param1: LDS v4i*; AS(3) value = low 32 bits of generic
  __builtin_amdgcn_global_load_async_to_lds_b128(
      (__attribute__((address_space(1))) v4i*)(uintptr_t)src,
      (__attribute__((address_space(3))) v4i*)(uintptr_t)dst,
      0, 0);
#else
  *(uint4*)dst = *(const uint4*)src;
#endif
}
__device__ __forceinline__ void wait_async(){
#if defined(__gfx1250__) && __has_builtin(__builtin_amdgcn_global_load_async_to_lds_b128)
 #if __has_builtin(__builtin_amdgcn_s_wait_asynccnt)
  __builtin_amdgcn_s_wait_asynccnt(0);
 #else
  asm volatile("s_wait_asynccnt 0" ::: "memory");
 #endif
#endif
}

// ---- f32 -> f16 caching kernels -------------------------------------------
__global__ void k_cvt(const float* __restrict__ s, _Float16* __restrict__ d, int n){
  int i = blockIdx.x*256 + threadIdx.x;
  if (i < n) d[i] = (_Float16)s[i];
}
__global__ void k_cvtT128(const float* __restrict__ s, _Float16* __restrict__ d){
  int i = blockIdx.x*256 + threadIdx.x;   // 128x128 transpose
  int e = i>>7, dd = i&127;
  d[e*128+dd] = (_Float16)s[dd*128+e];
}

// ---- 1) fused tagger: softmax(tokens@tw@ext^T) with online softmax --------
__global__ __launch_bounds__(256) void k_tagger(
    const float* __restrict__ tokens, const _Float16* __restrict__ vocab_h,
    const _Float16* __restrict__ twT_h, _Float16* __restrict__ tagged_h)
{
  __shared__ _Float16 sTok[16*128];
  __shared__ _Float16 sM[16*128];
  __shared__ _Float16 sP[16*128];
  __shared__ float redA[8][16];
  __shared__ float redB[8][16];
  __shared__ float rowM[16], rowS[16], rowSc[16], rowPL[16], lastE[16];

  const int tid = threadIdx.x, lane = tid & 31, w = tid >> 5;
  const int row0 = blockIdx.x * 16;
  const int lc = lane & 15, rb = (lane>>4)*8;
  const int myd = w*16 + lc;

  for (int i = tid; i < 16*128; i += 256)
    sTok[i] = (_Float16)tokens[row0*128 + i];
  if (tid < 16){ rowM[tid] = -3.0e38f; rowS[tid] = 0.f; rowPL[tid] = 0.f; }
  __syncthreads();

  { // M = tokens @ tagger_weight   (B rows = out col e, via cached twT)
    v8f c = {};
    #pragma unroll
    for (int k0 = 0; k0 < 128; k0 += 32){
      v16h af = fragA_s(sTok + k0, 128, lane);
      v16h bf = fragA_s(twT_h + (w*16)*128 + k0, 128, lane);
      c = WMMA(af, bf, c);
    }
    #pragma unroll
    for (int j=0;j<8;++j) sM[(rb+j)*128 + myd] = (_Float16)c[j];
  }
  __syncthreads();

  v8f acc = {};
  for (int c0 = 0; c0 < 2048; c0 += 128){
    v8f lg = {};
    #pragma unroll
    for (int k0=0;k0<128;k0+=32){
      v16h af = fragA_s(sM + k0, 128, lane);
      v16h bf = frag_ext(vocab_h, c0 + w*16, k0, lane);
      lg = WMMA(af, bf, lg);
    }
    int vcol = c0 + w*16 + lc;
    if (vcol > V_){
      #pragma unroll
      for (int j=0;j<8;++j) lg[j] = NEGV;
    }
    #pragma unroll
    for (int j=0;j<8;++j){
      float m = redmax16(lg[j]);
      if (lc==0) redA[w][rb+j] = m;
    }
    __syncthreads();
    if (tid < 16){
      float cm = redA[0][tid];
      #pragma unroll
      for (int q=1;q<8;++q) cm = fmaxf(cm, redA[q][tid]);
      float nm = fmaxf(rowM[tid], cm);
      rowSc[tid] = __expf(rowM[tid] - nm);
      rowM[tid]  = nm;
      lastE[tid] = 0.f;
    }
    __syncthreads();
    #pragma unroll
    for (int j=0;j<8;++j){
      int r = rb+j;
      float e = __expf(lg[j] - rowM[r]);
      sP[r*128 + w*16 + lc] = (_Float16)e;
      if (vcol == V_) lastE[r] = e;
      float s = redsum16(e);
      if (lc==0) redB[w][r] = s;
      acc[j] *= rowSc[r];
    }
    __syncthreads();
    if (tid < 16){
      float ts = 0.f;
      #pragma unroll
      for (int q=0;q<8;++q) ts += redB[q][tid];
      rowS[tid]  = rowS[tid]*rowSc[tid] + ts;
      rowPL[tid] = rowPL[tid]*rowSc[tid] + lastE[tid];
    }
    __syncthreads();
    #pragma unroll
    for (int k0=0;k0<128;k0+=32){
      v16h af = fragA_s(sP + k0, 128, lane);
      v16h bf = frag_vocT(vocab_h, c0 + k0, myd, lane);
      acc = WMMA(af, bf, acc);
    }
    __syncthreads();
  }
  #pragma unroll
  for (int j=0;j<8;++j){
    int r = rb+j;
    float inv = 1.f / rowS[r];
    float val = acc[j]*inv + (rowPL[r]*inv) * tokens[(row0+r)*128 + myd];
    tagged_h[(row0+r)*128 + myd] = (_Float16)val;
  }
}

// ---- 2) LSTM step ---------------------------------------------------------
__global__ __launch_bounds__(256) void k_lstm(
    const _Float16* __restrict__ tagged_h, const _Float16* __restrict__ Wih_h,
    const _Float16* __restrict__ Whh_h, const float* __restrict__ bih,
    const float* __restrict__ bhh, float* __restrict__ hstate,
    float* __restrict__ cstate, float* __restrict__ hs, int t)
{
  __shared__ _Float16 sX[16*256];
  __shared__ float sG[16*512];
  const int tid=threadIdx.x, lane=tid&31, w=tid>>5;
  const int b0 = blockIdx.x*16;
  for (int i=tid;i<16*256;i+=256){
    int r=i>>8, d=i&255;
    sX[i] = (d<128) ? tagged_h[((b0+r)*L_ + t)*128 + d]
                    : (_Float16)hstate[(b0+r)*128 + (d-128)];
  }
  __syncthreads();
  for (int q=0;q<4;++q){
    int col0 = (w + 8*q)*16;
    v8f c={};
    #pragma unroll
    for (int k0=0;k0<256;k0+=32){
      v16h af = fragA_s(sX + k0, 256, lane);
      v16h bf = (k0<128) ? fragA_s(Wih_h + col0*128 + k0, 128, lane)
                         : fragA_s(Whh_h + col0*128 + (k0-128), 128, lane);
      c = WMMA(af, bf, c);
    }
    int col = col0 + (lane&15), rbq=(lane>>4)*8;
    #pragma unroll
    for (int j=0;j<8;++j) sG[(rbq+j)*512 + col] = c[j] + bih[col] + bhh[col];
  }
  __syncthreads();
  for (int i=tid;i<16*128;i+=256){
    int r=i>>7, d=i&127, br=b0+r;
    float ig=sG[r*512+d],     fg=sG[r*512+128+d];
    float gg=sG[r*512+256+d], og=sG[r*512+384+d];
    float cN = sigf(fg)*cstate[br*128+d] + sigf(ig)*tanhf(gg);
    float h  = sigf(og)*tanhf(cN);
    cstate[br*128+d]=cN; hstate[br*128+d]=h;
    hs[t*B_*128 + br*128 + d] = h;
  }
}

__global__ void k_encoded(const float* __restrict__ hs, const int* __restrict__ lengths,
                          float* __restrict__ encoded){
  int b=blockIdx.x, d=threadIdx.x;
  int t=lengths[b]-1;
  encoded[b*128+d] = hs[t*B_*128 + b*128 + d];
}

// ---- 3) decoder -----------------------------------------------------------
__global__ __launch_bounds__(256) void k_dec_pre(
    const float* __restrict__ enc, const _Float16* __restrict__ Wdih_h,
    const float* __restrict__ bdih, float* __restrict__ pre){
  __shared__ _Float16 sX[16*128];
  const int tid=threadIdx.x, lane=tid&31, w=tid>>5;
  const int b0=blockIdx.x*16;
  for (int i=tid;i<16*128;i+=256) sX[i]=(_Float16)enc[b0*128+i];
  __syncthreads();
  v8f c={};
  #pragma unroll
  for (int k0=0;k0<128;k0+=32){
    v16h af=fragA_s(sX+k0,128,lane);
    v16h bf=fragA_s(Wdih_h + (w*16)*128 + k0, 128, lane);
    c=WMMA(af,bf,c);
  }
  int col=w*16+(lane&15), rb=(lane>>4)*8;
  #pragma unroll
  for (int j=0;j<8;++j) pre[(b0+rb+j)*128+col] = c[j] + bdih[col];
}

__global__ __launch_bounds__(256) void k_dec_step(
    const float* __restrict__ pre, const _Float16* __restrict__ Wdhh_h,
    const float* __restrict__ bdhh, float* __restrict__ hdec,
    float* __restrict__ hidden, int s){
  __shared__ _Float16 sX[16*128];
  const int tid=threadIdx.x, lane=tid&31, w=tid>>5;
  const int b0=blockIdx.x*16;
  for (int i=tid;i<16*128;i+=256) sX[i]=(_Float16)hdec[b0*128+i];
  __syncthreads();
  v8f c={};
  #pragma unroll
  for (int k0=0;k0<128;k0+=32){
    v16h af=fragA_s(sX+k0,128,lane);
    v16h bf=fragA_s(Wdhh_h + (w*16)*128 + k0, 128, lane);
    c=WMMA(af,bf,c);
  }
  __syncthreads();
  int col=w*16+(lane&15), rb=(lane>>4)*8;
  #pragma unroll
  for (int j=0;j<8;++j){
    int br=b0+rb+j;
    float v = fmaxf(pre[br*128+col] + c[j] + bdhh[col], 0.f);
    hdec[br*128+col]=v;
    hidden[(br*8+s)*128+col]=v;
  }
}

// ---- 4) attention over tokens (all 8 steps at once) -----------------------
__global__ __launch_bounds__(256) void k_attn(
    const float* __restrict__ hidden, const _Float16* __restrict__ tagged_h,
    const int* __restrict__ lengths, float* __restrict__ instructions){
  __shared__ _Float16 sH[8][16*128];
  __shared__ _Float16 sP[8][16*32];
  const int tid=threadIdx.x, lane=tid&31, w=tid>>5;
  const int b = blockIdx.x*8 + w;
  for (int i=lane;i<16*128;i+=32){
    int r=i>>7, d=i&127;
    sH[w][i] = (r<8) ? (_Float16)hidden[(b*8+r)*128+d] : (_Float16)0.f;
  }
  __syncthreads();
  const int len = lengths[b];
  v8f s0={}, s1={};
  #pragma unroll
  for (int k0=0;k0<128;k0+=32){
    v16h af = fragA_s(sH[w]+k0, 128, lane);
    v16h b0 = fragA_s(tagged_h + (b*L_ + 0 )*128 + k0, 128, lane);
    v16h b1 = fragA_s(tagged_h + (b*L_ + 16)*128 + k0, 128, lane);
    s0 = WMMA(af, b0, s0);
    s1 = WMMA(af, b1, s1);
  }
  const int lc=lane&15, rb=(lane>>4)*8;
  #pragma unroll
  for (int j=0;j<8;++j){
    float v0 = (lc      < len) ? s0[j] : NEGV;
    float v1 = (16 + lc < len) ? s1[j] : NEGV;
    float m  = redmax16(fmaxf(v0,v1));
    float e0 = __expf(v0-m), e1 = __expf(v1-m);
    float ss = redsum16(e0+e1);
    sP[w][(rb+j)*32 + lc]      = (_Float16)(e0/ss);
    sP[w][(rb+j)*32 + 16 + lc] = (_Float16)(e1/ss);
  }
  __syncthreads();
  v16h af = fragA_s(sP[w], 32, lane);
  for (int dt=0; dt<8; ++dt){
    v16h bf = fragB_th(tagged_h + b*L_*128 + dt*16, 128, lane);
    v8f c = {};
    c = WMMA(af, bf, c);
    #pragma unroll
    for (int j=0;j<8;++j){
      int row = rb+j;
      if (row < 8) instructions[(b*8+row)*128 + dt*16 + lc] = c[j];
    }
  }
}

// ---- 5) graph reasoning per-step kernels ----------------------------------
__global__ void k_propsim(const float* __restrict__ instructions,
                          const float* __restrict__ prop_embeds,
                          float* __restrict__ ps, int s){
  const int b=blockIdx.x, lane=threadIdx.x; // 32 threads
  float acc[8]={0,0,0,0,0,0,0,0};
  const float* iv = instructions + (b*8+s)*128;
  for (int d=lane; d<128; d+=32){
    float x = iv[d];
    #pragma unroll
    for (int p=0;p<8;++p) acc[p] += x * prop_embeds[p*128+d];
  }
  #pragma unroll
  for (int p=0;p<8;++p){
    #pragma unroll
    for (int m=1;m<32;m<<=1) acc[p] += __shfl_xor(acc[p], m, 32);
  }
  if (lane==0){
    float mx = acc[0];
    #pragma unroll
    for (int p=1;p<8;++p) mx = fmaxf(mx, acc[p]);
    float e[8], sm=0.f;
    #pragma unroll
    for (int p=0;p<8;++p){ e[p]=__expf(acc[p]-mx); sm+=e[p]; }
    #pragma unroll
    for (int p=0;p<8;++p) ps[b*8+p] = e[p]/sm;
  }
}

__global__ __launch_bounds__(256) void k_node(
    const _Float16* __restrict__ nattr_h, const _Float16* __restrict__ Wsp_h,
    const float* __restrict__ ps, const float* __restrict__ instructions,
    const float* __restrict__ W_state, const int* __restrict__ node_graph,
    float* __restrict__ nscore, int s){
  __shared__ _Float16 sA[16*128];
  __shared__ float red[16][8];
  const int tid=threadIdx.x, lane=tid&31, w=tid>>5;
  const int base = blockIdx.x*16;
  const int g = node_graph[base];
  const int cr = tid>>4, cq = tid&15;   // staging: row, 16B-chunk
  float acc8[8]={0,0,0,0,0,0,0,0};
  for (int p=0;p<7;++p){
    __syncthreads();
    ((uint4*)(sA + cr*128))[cq] =
        ((const uint4*)(nattr_h + ((base+cr)*7 + p)*128))[cq];
    if (p < 6)
      __builtin_prefetch(nattr_h + ((base+cr)*7 + p+1)*128 + cq*8, 0, 1);
    __syncthreads();
    v8f c = {};
    #pragma unroll
    for (int k0=0;k0<128;k0+=32){
      v16h af = fragA_s(sA+k0, 128, lane);
      v16h bf = fragA_s(Wsp_h + p*(128*128) + (w*16)*128 + k0, 128, lane);
      c = WMMA(af, bf, c);
    }
    float sc = ps[g*8+p];
    #pragma unroll
    for (int j=0;j<8;++j) acc8[j] += sc * c[j];
  }
  const int d = w*16 + (lane&15), rb=(lane>>4)*8;
  const float ivd = instructions[(g*8+s)*128 + d];
  const float wsd = W_state[d];
  #pragma unroll
  for (int j=0;j<8;++j){
    float v = eluf(acc8[j]*ivd) * wsd;
    float ss = redsum16(v);
    if ((lane&15)==0) red[rb+j][w] = ss;
  }
  __syncthreads();
  if (tid<16){
    float t=0.f;
    #pragma unroll
    for (int q=0;q<8;++q) t += red[tid][q];
    nscore[base+tid] = t;
  }
}

__global__ __launch_bounds__(256) void k_edge(
    const _Float16* __restrict__ eattr_h, const _Float16* __restrict__ Wsp7_h,
    const float* __restrict__ instructions, const float* __restrict__ W_rel,
    const int* __restrict__ edge_graph, const int* __restrict__ edge_src,
    const int* __restrict__ edge_dst, const float* __restrict__ dist,
    float* __restrict__ mscore, int s){
  __shared__ _Float16 sA[16*128];
  __shared__ float red[16][8];
  const int tid=threadIdx.x, lane=tid&31, w=tid>>5;
  const int base = blockIdx.x*16;
  const int g = edge_graph[base];
  // async global->LDS staging (pure f16 copy), 16B per thread
  lds_copy16_async(sA + tid*8, eattr_h + base*128 + tid*8);
  wait_async();
  __syncthreads();
  v8f c = {};
  #pragma unroll
  for (int k0=0;k0<128;k0+=32){
    v16h af = fragA_s(sA+k0, 128, lane);
    v16h bf = fragA_s(Wsp7_h + (w*16)*128 + k0, 128, lane);
    c = WMMA(af, bf, c);
  }
  const int d = w*16 + (lane&15), rb=(lane>>4)*8;
  const float ivd = instructions[(g*8+s)*128 + d];
  const float wrd = W_rel[d];
  #pragma unroll
  for (int j=0;j<8;++j){
    float v = eluf(c[j]*ivd) * wrd;
    float ss = redsum16(v);
    if ((lane&15)==0) red[rb+j][w] = ss;
  }
  __syncthreads();
  if (tid<16){
    float t=0.f;
    #pragma unroll
    for (int q=0;q<8;++q) t += red[tid][q];
    int e = base+tid;
    atomicAdd(&mscore[edge_dst[e]], dist[edge_src[e]] * t);
  }
}

__device__ float blk_max128(float v, float* sh){
  int tid=threadIdx.x; sh[tid]=v; __syncthreads();
  for (int off=64;off>0;off>>=1){
    if (tid<off) sh[tid]=fmaxf(sh[tid],sh[tid+off]);
    __syncthreads();
  }
  float r=sh[0]; __syncthreads(); return r;
}
__device__ float blk_sum128(float v, float* sh){
  int tid=threadIdx.x; sh[tid]=v; __syncthreads();
  for (int off=64;off>0;off>>=1){
    if (tid<off) sh[tid]=sh[tid]+sh[tid+off];
    __syncthreads();
  }
  float r=sh[0]; __syncthreads(); return r;
}

__global__ void k_update(const float* __restrict__ nscore, const float* __restrict__ mscore,
                         const float* __restrict__ ps, float* __restrict__ dist){
  __shared__ float sh[128];
  const int g=blockIdx.x, tid=threadIdx.x;
  const int n=g*128+tid;
  float a = nscore[n], m = mscore[n];
  float amax = blk_max128(a, sh);
  float ae = __expf(a-amax);
  float pa = ae / blk_sum128(ae, sh);
  float mmax = blk_max128(m, sh);
  float me = __expf(m-mmax);
  float pm = me / blk_sum128(me, sh);
  float r = ps[g*8+7];
  dist[n] = r*pm + (1.f-r)*pa;
}

__global__ void k_initdist(const int* __restrict__ node_graph,
                           const int* __restrict__ npg, float* __restrict__ dist){
  int n = blockIdx.x*256 + threadIdx.x;
  if (n < N_) dist[n] = 1.f/(float)npg[node_graph[n]];
}

__global__ void k_agg(const float* __restrict__ node_attrs, const float* __restrict__ ps,
                      const float* __restrict__ dist, float* __restrict__ agg){
  const int g=blockIdx.x, d=threadIdx.x; // 128 threads
  float p[7];
  #pragma unroll
  for (int q=0;q<7;++q) p[q]=ps[g*8+q];
  float acc=0.f;
  for (int nn=0;nn<NPG_;++nn){
    int node = g*NPG_ + nn;
    const float* a = node_attrs + node*7*128;
    float inner=0.f;
    #pragma unroll
    for (int q=0;q<7;++q) inner += p[q]*a[q*128+d];
    acc += dist[node]*inner;
  }
  agg[g*128+d]=acc;
}

// ---- 6) output GEMM: [encoded|agg] @ W_out^T + b_out ----------------------
__global__ __launch_bounds__(256) void k_out(
    const float* __restrict__ enc, const float* __restrict__ agg,
    const _Float16* __restrict__ Wout_h, const float* __restrict__ bout,
    float* __restrict__ out){
  __shared__ _Float16 sX[16*256];
  const int tid=threadIdx.x, lane=tid&31, w=tid>>5;
  const int b0=(blockIdx.x>>3)*16, cg=blockIdx.x&7;
  for (int i=tid;i<16*256;i+=256){
    int r=i>>8, d=i&255;
    float v = (d<128) ? enc[(b0+r)*128+d] : agg[(b0+r)*128+(d-128)];
    sX[i]=(_Float16)v;
  }
  __syncthreads();
  const int col0=(cg*8+w)*16;
  v8f c={};
  #pragma unroll
  for (int k0=0;k0<256;k0+=32){
    v16h af=fragA_s(sX+k0,256,lane);
    v16h bf=fragA_s(Wout_h + col0*256 + k0, 256, lane);
    c=WMMA(af,bf,c);
  }
  const int col=col0+(lane&15), rb=(lane>>4)*8;
  #pragma unroll
  for (int j=0;j<8;++j)
    out[(b0+rb+j)*OUT_+col] = c[j] + bout[col];
}

// ---- host -----------------------------------------------------------------
static inline int cdiv(long n, int b){ return (int)((n + b - 1) / b); }

extern "C" void kernel_launch(void* const* d_in, const int* in_sizes, int n_in,
                              void* d_out, int out_size, void* d_ws, size_t ws_size,
                              hipStream_t stream){
  (void)in_sizes; (void)n_in; (void)out_size; (void)ws_size;
  const float* tokens      = (const float*)d_in[0];
  const float* vocab       = (const float*)d_in[1];
  const float* prop_embeds = (const float*)d_in[2];
  const float* node_attrs  = (const float*)d_in[3];
  const float* edge_attrs  = (const float*)d_in[4];
  const float* defemb      = (const float*)d_in[5];
  const float* tw          = (const float*)d_in[6];
  const float* Wih         = (const float*)d_in[7];
  const float* Whh         = (const float*)d_in[8];
  const float* bih         = (const float*)d_in[9];
  const float* bhh         = (const float*)d_in[10];
  const float* Wdih        = (const float*)d_in[11];
  const float* Wdhh        = (const float*)d_in[12];
  const float* bdih        = (const float*)d_in[13];
  const float* bdhh        = (const float*)d_in[14];
  const float* Wsp         = (const float*)d_in[15];
  const float* Wstate      = (const float*)d_in[16];
  const float* Wrel        = (const float*)d_in[17];
  const float* Wout        = (const float*)d_in[18];
  const float* bout        = (const float*)d_in[19];
  const int*   node_graph  = (const int*)d_in[20];
  const int*   edge_graph  = (const int*)d_in[21];
  const int*   lengths     = (const int*)d_in[22];
  const int*   npg         = (const int*)d_in[23];
  const int*   edge_src    = (const int*)d_in[24];
  const int*   edge_dst    = (const int*)d_in[25];

  // f32 scratch
  float* ws = (float*)d_ws;
  float* hs           = ws; ws += L_*B_*D_;
  float* hstate       = ws; ws += B_*D_;
  float* cstate       = ws; ws += B_*D_;
  float* encoded      = ws; ws += B_*D_;
  float* pre          = ws; ws += B_*D_;
  float* hdec         = ws; ws += B_*D_;
  float* hidden       = ws; ws += B_*STEPS_*D_;
  float* instructions = ws; ws += B_*STEPS_*D_;
  float* ps           = ws; ws += B_*P_;
  float* nscore       = ws; ws += N_;
  float* mscore       = ws; ws += N_;
  float* dist         = ws; ws += N_;
  float* agg          = ws; ws += B_*D_;
  // f16 caches (16B-aligned: all preceding counts are multiples of 4 floats)
  _Float16* h = (_Float16*)ws;
  _Float16* vocab_h  = h; h += (V_+1)*D_;       // 2001 rows: vocab + default
  _Float16* twT_h    = h; h += D_*D_;
  _Float16* Wih_h    = h; h += 4*D_*D_;
  _Float16* Whh_h    = h; h += 4*D_*D_;
  _Float16* Wdih_h   = h; h += D_*D_;
  _Float16* Wdhh_h   = h; h += D_*D_;
  _Float16* Wsp_h    = h; h += P_*D_*D_;
  _Float16* Wout_h   = h; h += OUT_*2*D_;
  _Float16* tagged_h = h; h += B_*L_*D_;
  _Float16* nattr_h  = h; h += (long)N_*7*D_;
  _Float16* eattr_h  = h; h += (long)E_*D_;

  // one-time (per launch) f16 caching
  k_cvt<<<cdiv(V_*D_,256),256,0,stream>>>(vocab, vocab_h, V_*D_);
  k_cvt<<<1,256,0,stream>>>(defemb, vocab_h + V_*D_, D_);
  k_cvtT128<<<64,256,0,stream>>>(tw, twT_h);
  k_cvt<<<cdiv(4*D_*D_,256),256,0,stream>>>(Wih, Wih_h, 4*D_*D_);
  k_cvt<<<cdiv(4*D_*D_,256),256,0,stream>>>(Whh, Whh_h, 4*D_*D_);
  k_cvt<<<cdiv(D_*D_,256),256,0,stream>>>(Wdih, Wdih_h, D_*D_);
  k_cvt<<<cdiv(D_*D_,256),256,0,stream>>>(Wdhh, Wdhh_h, D_*D_);
  k_cvt<<<cdiv(P_*D_*D_,256),256,0,stream>>>(Wsp, Wsp_h, P_*D_*D_);
  k_cvt<<<cdiv(OUT_*2*D_,256),256,0,stream>>>(Wout, Wout_h, OUT_*2*D_);
  k_cvt<<<cdiv((long)N_*7*D_,256),256,0,stream>>>(node_attrs, nattr_h, N_*7*D_);
  k_cvt<<<cdiv((long)E_*D_,256),256,0,stream>>>(edge_attrs, eattr_h, E_*D_);

  k_tagger<<<(B_*L_)/16, 256, 0, stream>>>(tokens, vocab_h, twT_h, tagged_h);

  (void)hipMemsetAsync(hstate, 0, B_*D_*sizeof(float), stream);
  (void)hipMemsetAsync(cstate, 0, B_*D_*sizeof(float), stream);
  for (int t=0;t<L_;++t)
    k_lstm<<<B_/16, 256, 0, stream>>>(tagged_h, Wih_h, Whh_h, bih, bhh, hstate, cstate, hs, t);
  k_encoded<<<B_, 128, 0, stream>>>(hs, lengths, encoded);

  k_dec_pre<<<B_/16, 256, 0, stream>>>(encoded, Wdih_h, bdih, pre);
  (void)hipMemsetAsync(hdec, 0, B_*D_*sizeof(float), stream);
  for (int s=0;s<STEPS_;++s)
    k_dec_step<<<B_/16, 256, 0, stream>>>(pre, Wdhh_h, bdhh, hdec, hidden, s);

  k_attn<<<B_/8, 256, 0, stream>>>(hidden, tagged_h, lengths, instructions);

  k_initdist<<<N_/256, 256, 0, stream>>>(node_graph, npg, dist);
  for (int s=0;s<STEPS_;++s){
    k_propsim<<<B_, 32, 0, stream>>>(instructions, prop_embeds, ps, s);
    k_node<<<N_/16, 256, 0, stream>>>(nattr_h, Wsp_h, ps, instructions, Wstate,
                                      node_graph, nscore, s);
    (void)hipMemsetAsync(mscore, 0, N_*sizeof(float), stream);
    k_edge<<<E_/16, 256, 0, stream>>>(eattr_h, Wsp_h + 7*D_*D_, instructions, Wrel,
                                      edge_graph, edge_src, edge_dst, dist, mscore, s);
    k_update<<<B_, 128, 0, stream>>>(nscore, mscore, ps, dist);
  }
  k_agg<<<B_, 128, 0, stream>>>(node_attrs, ps, dist, agg);
  k_out<<<(B_/16)*8, 256, 0, stream>>>(encoded, agg, Wout_h, bout, (float*)d_out);
}
